// MambaActorCritic_77086073028641
// MI455X (gfx1250) — matrix operand order repaired and checked by reading, not compiled, not run on hardware
//
#include <hip/hip_runtime.h>
#include <hip/hip_bf16.h>
#include <stdint.h>

// ---------------------------------------------------------------------------
// Mamba actor-critic forward for MI455X (gfx1250, wave32).
// Dense projections -> v_wmma_f32_16x16x32_bf16 (bf16 in, f32 accum),
//   16x64 output strip per wave (4 accumulators, A-fragment reuse).
// Selective scan   -> 1 thread per (b,d,n), 16-lane shfl_xor reduction,
//   dbc staged to LDS via global_load_async_to_lds_b32 (ASYNCcnt path).
// ---------------------------------------------------------------------------

#define BATCH     4
#define SEQLEN    1024
#define OBS_DIM   60
#define ACT_DIM   2
#define D_MODEL   256
#define N_LAYERS  4
#define D_STATE   16
#define D_CONV    4
#define D_INNER   512
#define DT_RANK   16
#define BL        (BATCH * SEQLEN)          // 4096 rows
#define SCAN_CHUNK 8

typedef __attribute__((ext_vector_type(16))) __bf16 v16bf;
typedef __attribute__((ext_vector_type(8)))  float  v8f;

__device__ __forceinline__ float sigmoidf_(float x) {
  return 1.0f / (1.0f + __expf(-x));
}

// ---------------------------------------------------------------------------
// WMMA fragment loaders (wave32; A/B use contiguous k = 16*(lane/16) + s, same in both, not the ISA table order).
// A: 16x32 bf16, row M = lane&15; K base = (lane&16)?16:0;
//    VGPR v holds K = kb + 2v, packed pairs.
// B: 32x16 bf16, col N = lane&15; K base = (lane&16)?16:0; VGPR v: K = kb+2v.
// C/D: VGPR r -> row = r + ((lane&16)?8:0), col = lane&15.
// ---------------------------------------------------------------------------
__device__ __forceinline__ v16bf load_a_frag(const float* __restrict__ A,
                                             int lda, int m0, int kc, int lane) {
  const float* p = A + (size_t)(m0 + (lane & 15)) * lda;
  int kb = kc + ((lane & 16) ? 16 : 0);
  v16bf f;
#pragma unroll
  for (int v = 0; v < 8; ++v) {
    int k = kb + 2 * v;
    f[2 * v]     = (__bf16)p[k];
    f[2 * v + 1] = (__bf16)p[k + 1];
  }
  return f;
}

__device__ __forceinline__ v16bf load_b_frag(const float* __restrict__ W,
                                             int ldw, int n0, int kc, int lane) {
  const float* p = W + (size_t)(n0 + (lane & 15)) * ldw;  // W row-major (N,K)
  int kb = kc + ((lane & 16) ? 16 : 0);
  v16bf f;
#pragma unroll
  for (int v = 0; v < 8; ++v) {
    int k = kb + 2 * v;
    f[2 * v]     = (__bf16)p[k];
    f[2 * v + 1] = (__bf16)p[k + 1];
  }
  return f;
}

__device__ __forceinline__ v8f wmma_bf16(const v16bf& a, const v16bf& b, const v8f& c) {
  return __builtin_amdgcn_wmma_f32_16x16x32_bf16(
      /*neg_a=*/false, a, /*neg_b=*/false, b,
      /*c_mod=*/(short)0, c, /*reuse_a=*/false, /*reuse_b=*/false);
}

template <bool RESID>
__device__ __forceinline__ void store_tile(float* __restrict__ out,
                                           const float* __restrict__ resid,
                                           int N, int m0, int n0, int lane,
                                           const v8f& c) {
  int col   = n0 + (lane & 15);
  int rbase = m0 + ((lane & 16) ? 8 : 0);
#pragma unroll
  for (int r = 0; r < 8; ++r) {
    size_t idx = (size_t)(rbase + r) * N + col;
    float v = c[r];
    if (RESID) v += resid[idx];
    out[idx] = v;
  }
}

// out[M,N] = A[M,K] * W[N,K]^T (+ resid).
// One wave owns a 16x64 strip (4 WMMA accumulators, shared A fragment).
// 4 waves per block -> block covers 16x256 of the output.
template <bool RESID>
__global__ void gemm_wmma_kernel(const float* __restrict__ A,
                                 const float* __restrict__ W,
                                 const float* __restrict__ resid,
                                 float* __restrict__ out,
                                 int M, int N, int K) {
  int lane  = threadIdx.x & 31;
  int wave  = threadIdx.x >> 5;
  int m0    = blockIdx.x * 16;
  int nbase = (blockIdx.y * 4 + wave) * 64;
  if (m0 >= M || nbase >= N) return;        // wave-uniform: EXEC stays all-ones

  int nact = (N - nbase) >> 4;              // 16-wide tiles in range (N % 16 == 0)
  if (nact > 4) nact = 4;

  v8f c0 = {}, c1 = {}, c2 = {}, c3 = {};
  const float* arow = A + (size_t)(m0 + (lane & 15)) * K;
  for (int kc = 0; kc < K; kc += 32) {
    if (kc + 32 < K)                        // prefetch next A chunk -> global_prefetch_b8
      __builtin_prefetch(arow + kc + 32, 0, 3);
    v16bf a = load_a_frag(A, K, m0, kc, lane);
    v16bf b = load_b_frag(W, K, nbase, kc, lane);
    c0 = wmma_bf16(a, b, c0);
    if (nact > 1) { b = load_b_frag(W, K, nbase + 16, kc, lane); c1 = wmma_bf16(a, b, c1); }
    if (nact > 2) { b = load_b_frag(W, K, nbase + 32, kc, lane); c2 = wmma_bf16(a, b, c2); }
    if (nact > 3) { b = load_b_frag(W, K, nbase + 48, kc, lane); c3 = wmma_bf16(a, b, c3); }
  }

  store_tile<RESID>(out, resid, N, m0, nbase, lane, c0);
  if (nact > 1) store_tile<RESID>(out, resid, N, m0, nbase + 16, lane, c1);
  if (nact > 2) store_tile<RESID>(out, resid, N, m0, nbase + 32, lane, c2);
  if (nact > 3) store_tile<RESID>(out, resid, N, m0, nbase + 48, lane, c3);
}

// ---------------------------------------------------------------------------
// x = silu(obs @ Win^T + bin)   (K=60, tiny -> VALU)
// ---------------------------------------------------------------------------
__global__ void input_proj_kernel(const float* __restrict__ obs,
                                  const float* __restrict__ w,
                                  const float* __restrict__ bias,
                                  float* __restrict__ x) {
  int idx = blockIdx.x * blockDim.x + threadIdx.x;
  if (idx >= BL * D_MODEL) return;
  int row = idx >> 8;           // / D_MODEL
  int d   = idx & (D_MODEL - 1);
  const float* o  = obs + (size_t)row * OBS_DIM;
  const float* wr = w + (size_t)d * OBS_DIM;
  float acc = bias[d];
#pragma unroll 4
  for (int k = 0; k < OBS_DIM; ++k) acc = fmaf(o[k], wr[k], acc);
  x[idx] = acc * sigmoidf_(acc);
}

// ---------------------------------------------------------------------------
// LayerNorm over D_MODEL=256, one block per row.
// ---------------------------------------------------------------------------
__global__ void layernorm_kernel(const float* __restrict__ x,
                                 const float* __restrict__ w,
                                 const float* __restrict__ b,
                                 float* __restrict__ xn) {
  __shared__ float red[D_MODEL];
  int row = blockIdx.x;
  int t   = threadIdx.x;
  float v = x[(size_t)row * D_MODEL + t];
  red[t] = v;
  __syncthreads();
  for (int s = D_MODEL / 2; s > 0; s >>= 1) {
    if (t < s) red[t] += red[t + s];
    __syncthreads();
  }
  float mean = red[0] * (1.0f / D_MODEL);
  __syncthreads();
  float dv = v - mean;
  red[t] = dv * dv;
  __syncthreads();
  for (int s = D_MODEL / 2; s > 0; s >>= 1) {
    if (t < s) red[t] += red[t + s];
    __syncthreads();
  }
  float r = rsqrtf(red[0] * (1.0f / D_MODEL) + 1e-5f);
  xn[(size_t)row * D_MODEL + t] = dv * r * w[t] + b[t];
}

// ---------------------------------------------------------------------------
// Causal depthwise conv (width 4) + bias + silu on xz[:, :, :D_INNER].
// ---------------------------------------------------------------------------
__global__ void conv_silu_kernel(const float* __restrict__ xz,
                                 const float* __restrict__ cw,
                                 const float* __restrict__ cb,
                                 float* __restrict__ xi) {
  int idx = blockIdx.x * blockDim.x + threadIdx.x;
  if (idx >= BATCH * SEQLEN * D_INNER) return;
  int d = idx & (D_INNER - 1);
  int l = (idx >> 9) & (SEQLEN - 1);
  int b = idx >> 19;
  float acc = cb[d];
#pragma unroll
  for (int k = 0; k < D_CONV; ++k) {
    int ls = l + k - (D_CONV - 1);
    if (ls >= 0)
      acc = fmaf(xz[((size_t)(b * SEQLEN + ls) * 2 * D_INNER) + d],
                 cw[d * D_CONV + k], acc);
  }
  xi[idx] = acc * sigmoidf_(acc);
}

// ---------------------------------------------------------------------------
// Selective scan. Grid (B, D_INNER/16), block 256 = 16 d x 16 n.
// Thread (dl,n): recurrence h = exp(dt*A)*h + dt*xi*B; y = sum_n h*C.
// n-reduction via 16-lane shfl_xor. Per SCAN_CHUNK=8 steps: dbc rows
// (contiguous, 8*48 floats) are staged to LDS with the gfx1250 async
// LDS-DMA path (global_load_async_to_lds_b32, waited with s_wait_asynccnt).
// ---------------------------------------------------------------------------
__global__ void scan_kernel(const float* __restrict__ dbc,
                            const float* __restrict__ xi,
                            const float* __restrict__ xz,
                            const float* __restrict__ dtw,
                            const float* __restrict__ dtb,
                            const float* __restrict__ a_log,
                            const float* __restrict__ dskip,
                            float* __restrict__ ygated) {
  __shared__ float s_dbc[SCAN_CHUNK * 48];
  __shared__ float s_xi[SCAN_CHUNK * 16];
  __shared__ float s_z[SCAN_CHUNK * 16];
  int b  = blockIdx.x;
  int dg = blockIdx.y;
  int t  = threadIdx.x;
  int n  = t & 15;
  int dl = t >> 4;                // 0..15
  int d  = dg * 16 + dl;

  float dtw_r[DT_RANK];
#pragma unroll
  for (int r = 0; r < DT_RANK; ++r) dtw_r[r] = dtw[d * DT_RANK + r];
  float dtb_d = dtb[d];
  float A_dn  = -__expf(a_log[d * D_STATE + n]);
  float Dd    = dskip[d];
  float h = 0.0f;

  for (int l0 = 0; l0 < SEQLEN; l0 += SCAN_CHUNK) {
    __syncthreads();   // previous chunk fully consumed before overwrite

    // dbc rows are contiguous: 8 rows * 48 floats = one contiguous burst.
    // Async DMA straight into LDS (no VGPR round trip), ASYNCcnt-tracked.
    const float* gdbc = dbc + (size_t)(b * SEQLEN + l0) * 48;
    for (int i = t; i < SCAN_CHUNK * 48; i += 256) {
      unsigned lds_off = (unsigned)(uintptr_t)(&s_dbc[i]);  // addr[31:0] = DS offset
      unsigned goff    = (unsigned)(i * 4);
      asm volatile("global_load_async_to_lds_b32 %0, %1, %2"
                   :: "v"(lds_off), "v"(goff), "s"(gdbc) : "memory");
    }
    // xi / z slices are strided; plain loads + ds_store.
    for (int i = t; i < SCAN_CHUNK * 16; i += 256) {
      int r = i >> 4, c = i & 15;
      size_t rowB = (size_t)(b * SEQLEN + l0 + r);
      s_xi[i] = xi[rowB * D_INNER + dg * 16 + c];
      s_z[i]  = xz[rowB * 2 * D_INNER + D_INNER + dg * 16 + c];
    }
    asm volatile("s_wait_asynccnt 0x0" ::: "memory");
    __syncthreads();

#pragma unroll
    for (int j = 0; j < SCAN_CHUNK; ++j) {
      const float* db = &s_dbc[j * 48];
      float dt_lin = dtb_d;
#pragma unroll
      for (int r = 0; r < DT_RANK; ++r) dt_lin = fmaf(db[r], dtw_r[r], dt_lin);
      float dt = (dt_lin > 20.0f) ? dt_lin : log1pf(__expf(dt_lin));

      float xiv = s_xi[j * 16 + dl];
      float Bn  = db[DT_RANK + n];
      float Cn  = db[DT_RANK + D_STATE + n];
      float dA  = __expf(dt * A_dn);
      h = fmaf(dA, h, dt * xiv * Bn);
      float contrib = h * Cn;
      contrib += __shfl_xor(contrib, 1, 16);
      contrib += __shfl_xor(contrib, 2, 16);
      contrib += __shfl_xor(contrib, 4, 16);
      contrib += __shfl_xor(contrib, 8, 16);
      if (n == 0) {
        float zv = s_z[j * 16 + dl];
        float y  = fmaf(Dd, xiv, contrib);
        ygated[(size_t)(b * SEQLEN + l0 + j) * D_INNER + d] = y * (zv * sigmoidf_(zv));
      }
    }
  }
}

// ---------------------------------------------------------------------------
// Heads: mu (4096x2), std (exp(log_std) broadcast), value (4096x1).
// ---------------------------------------------------------------------------
__global__ void heads_kernel(const float* __restrict__ x,
                             const float* __restrict__ pw,
                             const float* __restrict__ pb,
                             const float* __restrict__ plogstd,
                             const float* __restrict__ vw,
                             const float* __restrict__ vb,
                             float* __restrict__ out) {
  int row = blockIdx.x * blockDim.x + threadIdx.x;
  if (row >= BL) return;
  const float* xr = x + (size_t)row * D_MODEL;
  float m0 = pb[0], m1 = pb[1], val = vb[0];
#pragma unroll 4
  for (int k = 0; k < D_MODEL; ++k) {
    float xv = xr[k];
    m0  = fmaf(xv, pw[k], m0);
    m1  = fmaf(xv, pw[D_MODEL + k], m1);
    val = fmaf(xv, vw[k], val);
  }
  out[(size_t)row * 2 + 0] = m0;
  out[(size_t)row * 2 + 1] = m1;
  out[BL * 2 + (size_t)row * 2 + 0] = __expf(plogstd[0]);
  out[BL * 2 + (size_t)row * 2 + 1] = __expf(plogstd[1]);
  out[BL * 4 + row] = val;
}

// ---------------------------------------------------------------------------
extern "C" void kernel_launch(void* const* d_in, const int* in_sizes, int n_in,
                              void* d_out, int out_size, void* d_ws, size_t ws_size,
                              hipStream_t stream) {
  const float* obs      = (const float*)d_in[0];
  const float* in_pw    = (const float*)d_in[1];
  const float* in_pb    = (const float*)d_in[2];
  const float* norm_w   = (const float*)d_in[3];
  const float* norm_b   = (const float*)d_in[4];
  const float* inproj_w = (const float*)d_in[5];
  const float* conv_w   = (const float*)d_in[6];
  const float* conv_b   = (const float*)d_in[7];
  const float* xproj_w  = (const float*)d_in[8];
  const float* dtproj_w = (const float*)d_in[9];
  const float* dtproj_b = (const float*)d_in[10];
  const float* A_log    = (const float*)d_in[11];
  const float* D_skip   = (const float*)d_in[12];
  const float* outp_w   = (const float*)d_in[13];
  const float* pol_w    = (const float*)d_in[14];
  const float* pol_b    = (const float*)d_in[15];
  const float* pol_ls   = (const float*)d_in[16];
  const float* val_w    = (const float*)d_in[17];
  const float* val_b    = (const float*)d_in[18];
  float* out = (float*)d_out;

  // workspace layout (floats)
  float* ws  = (float*)d_ws;
  float* x   = ws;                              // BL*256      = 1,048,576
  float* xn  = x   + (size_t)BL * D_MODEL;      // BL*256
  float* xz  = xn  + (size_t)BL * D_MODEL;      // BL*1024     = 4,194,304
  float* xi  = xz  + (size_t)BL * 2 * D_INNER;  // BL*512      = 2,097,152
  float* dbc = xi  + (size_t)BL * D_INNER;      // BL*48       =   196,608
  float* yg  = dbc + (size_t)BL * 48;           // BL*512      = 2,097,152

  input_proj_kernel<<<(BL * D_MODEL + 255) / 256, 256, 0, stream>>>(
      obs, in_pw, in_pb, x);

  for (int layer = 0; layer < N_LAYERS; ++layer) {
    layernorm_kernel<<<BL, D_MODEL, 0, stream>>>(
        x, norm_w + layer * D_MODEL, norm_b + layer * D_MODEL, xn);

    // xz = xn @ in_proj_w^T          (4096x256 -> 4096x1024)
    gemm_wmma_kernel<false><<<dim3(BL / 16, (2 * D_INNER) / 256), 128, 0, stream>>>(
        xn, inproj_w + (size_t)layer * 2 * D_INNER * D_MODEL, nullptr, xz,
        BL, 2 * D_INNER, D_MODEL);

    conv_silu_kernel<<<(BATCH * SEQLEN * D_INNER + 255) / 256, 256, 0, stream>>>(
        xz, conv_w + layer * D_INNER * D_CONV, conv_b + layer * D_INNER, xi);

    // dbc = xi @ x_proj_w^T          (4096x512 -> 4096x48)
    gemm_wmma_kernel<false><<<dim3(BL / 16, 1), 128, 0, stream>>>(
        xi, xproj_w + (size_t)layer * 48 * D_INNER, nullptr, dbc,
        BL, 48, D_INNER);

    scan_kernel<<<dim3(BATCH, D_INNER / 16), 256, 0, stream>>>(
        dbc, xi, xz,
        dtproj_w + (size_t)layer * D_INNER * DT_RANK,
        dtproj_b + layer * D_INNER,
        A_log + (size_t)layer * D_INNER * D_STATE,
        D_skip + layer * D_INNER, yg);

    // x = x + yg @ out_proj_w^T      (4096x512 -> 4096x256, fused residual)
    gemm_wmma_kernel<true><<<dim3(BL / 16, 1), 128, 0, stream>>>(
        yg, outp_w + (size_t)layer * D_MODEL * D_INNER, x, x,
        BL, D_MODEL, D_INNER);
  }

  heads_kernel<<<(BL + 255) / 256, 256, 0, stream>>>(
      x, pol_w, pol_b, pol_ls, val_w, val_b, out);
}
